// MultiHeadAttention_20332375180051
// MI455X (gfx1250) — matrix operand order repaired and checked by reading, not compiled, not run on hardware
//
#include <hip/hip_runtime.h>
#include <hip/hip_bf16.h>
#include <stdint.h>

#define B_   4
#define T_   2048
#define HID_ 1024
#define H_   16
#define DK_  64
#define DV_  64

typedef __attribute__((ext_vector_type(16))) __bf16 v16bf;
typedef __attribute__((ext_vector_type(8)))  float  v8f;
typedef __attribute__((ext_vector_type(4)))  unsigned int v4u;
typedef __attribute__((ext_vector_type(8)))  int v8i;
typedef __attribute__((ext_vector_type(4)))  int v4i;

#define HAVE_TDM __has_builtin(__builtin_amdgcn_tensor_load_to_lds)

struct U32x8 { uint4 a, b; };

__device__ __forceinline__ __bf16 f2bf(float f) {
  uint32_t u = __builtin_bit_cast(uint32_t, f);
  uint32_t r = (u + 0x7FFFu + ((u >> 16) & 1u)) >> 16;  // round to nearest even
  return __builtin_bit_cast(__bf16, (uint16_t)r);
}

__device__ __forceinline__ v16bf pack16(const __bf16* p0, const __bf16* p1) {
  U32x8 t;
  t.a = *(const uint4*)p0;
  t.b = *(const uint4*)p1;
  return __builtin_bit_cast(v16bf, t);
}

// A tile (16xK=32) bf16, row-major source (ISA 7.12.2 A layout).
__device__ __forceinline__ v16bf ldA(const __bf16* base, int row0, int ld, int k0, int lane) {
  int hh = lane >> 4, r = lane & 15;
  const __bf16* p = base + (size_t)(row0 + r) * ld + k0 + 8 * hh;
  return pack16(p, p + 16);
}

// B tile (K=32 x N=16) given B^T row-major (row = output column, K contiguous).
__device__ __forceinline__ v16bf ldB(const __bf16* baseT, int col0, int ld, int k0, int lane) {
  int hh = lane >> 4, c = lane & 15;
  const __bf16* p = baseT + (size_t)(col0 + c) * ld + k0 + 8 * hh;
  return pack16(p, p + 16);
}

__device__ __forceinline__ v8f wmma_bf(v16bf a, v16bf b, v8f c) {
  return __builtin_amdgcn_wmma_f32_16x16x32_bf16(false, a, false, b, (short)0, c,
                                                 false, false);
}

__device__ __forceinline__ unsigned rfl(unsigned x) {
  return (unsigned)__builtin_amdgcn_readfirstlane((int)x);
}
__device__ __forceinline__ unsigned long long rfl64(unsigned long long x) {
  return ((unsigned long long)rfl((unsigned)(x >> 32)) << 32) | rfl((unsigned)x);
}

#if HAVE_TDM
__device__ __forceinline__ unsigned lds_offset_of(const void* p) {
  return (unsigned)(unsigned long long)
      (__attribute__((address_space(3))) char*)(char*)p;
}

// Loop-invariant part of the D# (group 1), ISA ch.8: data_size=2B, 2-D tile.
__device__ __forceinline__ v8i tdm_desc_g1(unsigned tile_w, unsigned tile_h,
                                           unsigned tensor_w, unsigned tensor_h,
                                           unsigned long long row_stride) {
  v8i g1;
  g1[0] = (int)(1u << 16);                                    // data_size=1 (2B)
  g1[1] = (int)((tensor_w & 0xFFFFu) << 16);                  // tensor_dim0 lo
  g1[2] = (int)((tensor_w >> 16) | ((tensor_h & 0xFFFFu) << 16));
  g1[3] = (int)((tensor_h >> 16) | (tile_w << 16));           // tile_dim0
  g1[4] = (int)(tile_h & 0xFFFFu);                            // tile_dim1
  g1[5] = (int)(unsigned)row_stride;                          // dim0_stride lo32
  g1[6] = (int)(unsigned)((row_stride >> 32) & 0xFFFFull);    // dim0_stride hi16
  g1[7] = 0;
  return g1;
}

__device__ __forceinline__ void tdm_issue(v8i g1, unsigned lds_off,
                                          unsigned long long ga) {
  v4u g0 = { 1u,                                   // count=1, user mode
             lds_off,                              // lds byte address
             (unsigned)ga,                         // global addr lo
             (unsigned)((ga >> 32) & 0x1FFFFFFull) | (2u << 30) };  // hi + type=2
  v4i gz = {0, 0, 0, 0};
#if defined(__clang_major__) && __clang_major__ >= 23
  v8i gz8 = {0, 0, 0, 0, 0, 0, 0, 0};
  __builtin_amdgcn_tensor_load_to_lds(g0, g1, gz, gz, gz8, 0);
#else
  __builtin_amdgcn_tensor_load_to_lds(g0, g1, gz, gz, 0);
#endif
}
#endif

// ---------------- fp32 -> bf16 conversion ----------------
__global__ void cvt_kernel(const float* __restrict__ src, __bf16* __restrict__ dst, int n) {
  for (int i = blockIdx.x * blockDim.x + threadIdx.x; i < n; i += gridDim.x * blockDim.x)
    dst[i] = f2bf(src[i]);
}

// ---------------- QKV projection: Y = X @ W^T + bias ----------------
// One wave computes a 32x64 output tile (2x4 WMMA tiles): 12 b128 loads / 8 WMMAs.
__global__ __launch_bounds__(256) void qkv_proj_kernel(
    const __bf16* __restrict__ Xq, const __bf16* __restrict__ Xk, const __bf16* __restrict__ Xv,
    const __bf16* __restrict__ Wq, const __bf16* __restrict__ Wk, const __bf16* __restrict__ Wv,
    const float* __restrict__ bq, const float* __restrict__ bk, const float* __restrict__ bv,
    __bf16* __restrict__ Qd, __bf16* __restrict__ Kd, __bf16* __restrict__ VTd) {
  int lane = threadIdx.x & 31, wv = threadIdx.x >> 5;
  int z = blockIdx.z;
  const __bf16* X = (z == 0) ? Xq : (z == 1) ? Xk : Xv;
  const __bf16* W = (z == 0) ? Wq : (z == 1) ? Wk : Wv;
  const float* bias = (z == 0) ? bq : (z == 1) ? bk : bv;
  int g = blockIdx.x * 8 + wv;            // 4096 waves: 256 m-tiles x 16 n-groups
  int m0 = (g & 255) * 32;
  int n0 = (g >> 8) * 64;
  v8f acc[2][4] = {};
  for (int k0 = 0; k0 < HID_; k0 += 32) {
    v16bf a0 = ldA(X, m0, HID_, k0, lane);
    v16bf a1 = ldA(X, m0 + 16, HID_, k0, lane);
#pragma unroll
    for (int j = 0; j < 4; ++j) {
      v16bf bm = ldB(W, n0 + 16 * j, HID_, k0, lane);
      acc[0][j] = wmma_bf(a0, bm, acc[0][j]);
      acc[1][j] = wmma_bf(a1, bm, acc[1][j]);
    }
  }
  int col = lane & 15, hh = lane >> 4;
  if (z < 2) {
    __bf16* D = (z == 0) ? Qd : Kd;
#pragma unroll
    for (int j = 0; j < 4; ++j) {
      int n = n0 + 16 * j + col;
      float bn = bias[n];
      int h = n >> 6, d = n & 63;
#pragma unroll
      for (int i = 0; i < 2; ++i)
#pragma unroll
        for (int r = 0; r < 8; ++r) {
          int m = m0 + 16 * i + r + 8 * hh;
          int b = m >> 11, t = m & (T_ - 1);
          D[(((size_t)(b * H_ + h) * T_ + t) << 6) + d] = f2bf(acc[i][j][r] + bn);
        }
    }
  } else {
#pragma unroll
    for (int j = 0; j < 4; ++j) {
      int n = n0 + 16 * j + col;
      float bn = bias[n];
      int h = n >> 6, d = n & 63;
#pragma unroll
      for (int i = 0; i < 2; ++i)
#pragma unroll
        for (int r = 0; r < 8; ++r) {
          int m = m0 + 16 * i + r + 8 * hh;
          int b = m >> 11, t = m & (T_ - 1);
          VTd[(((size_t)(b * H_ + h) * 64 + d) << 11) + t] = f2bf(acc[i][j][r] + bn);
        }
    }
  }
}

// ---------------- Flash attention: one 16-query tile per wave ----------------
// K/V chunks staged into LDS by the Tensor Data Mover, double-buffered.
__global__ __launch_bounds__(256) void attn_kernel(
    const __bf16* __restrict__ Q, const __bf16* __restrict__ K, const __bf16* __restrict__ V,
    const int* __restrict__ mask, __bf16* __restrict__ O) {
  __shared__ __align__(16) __bf16 pbuf[8][16 * 32];
#if HAVE_TDM
  __shared__ __align__(16) __bf16 kbuf[8][2][32 * 64];   // [key][dk] row-major
  __shared__ __align__(16) __bf16 vbuf[8][2][64 * 32];   // [dv][key] row-major
#endif
  int lane = threadIdx.x & 31, wv = threadIdx.x >> 5;
  int g = blockIdx.x * 8 + wv;
  int qt = g & 127, h = (g >> 7) & 15, b = g >> 11;
  const __bf16* qbase = Q + ((size_t)(b * H_ + h) * T_) * DK_;
  const __bf16* kbase = K + ((size_t)(b * H_ + h) * T_) * DK_;
  const __bf16* vbase = V + ((size_t)(b * H_ + h) * DV_) * T_;  // [dv][t]
  const int* mrow = mask + b * T_;
  int q0 = qt * 16;
  int col = lane & 15, hh = lane >> 4;
  __bf16* pb = pbuf[wv];

  v16bf qa0 = ldA(qbase, q0, DK_, 0, lane);
  v16bf qa1 = ldA(qbase, q0, DK_, 32, lane);
  v8f o0 = {}, o1 = {}, o2 = {}, o3 = {};
  float rmax[8], rsum[8];
#pragma unroll
  for (int r = 0; r < 8; ++r) { rmax[r] = -1e30f; rsum[r] = 0.f; }

#if HAVE_TDM
  // Loop-invariant D# group1 + uniformized addresses/LDS offsets (SGPR-resident)
  v8i g1k = tdm_desc_g1(64, 32, 64, T_, 64);       // K: 32 keys x 64 dk
  v8i g1v = tdm_desc_g1(32, 64, T_, DV_, T_);      // V^T: 64 dv x 32 t
  unsigned kl[2] = { rfl(lds_offset_of(kbuf[wv][0])), rfl(lds_offset_of(kbuf[wv][1])) };
  unsigned vl[2] = { rfl(lds_offset_of(vbuf[wv][0])), rfl(lds_offset_of(vbuf[wv][1])) };
  unsigned long long kga = rfl64((unsigned long long)kbase);
  unsigned long long vga = rfl64((unsigned long long)vbase);
  tdm_issue(g1k, kl[0], kga);                      // prologue: chunk 0 -> buffer 0
  tdm_issue(g1v, vl[0], vga);
#endif

  for (int kt = 0; kt < T_; kt += 32) {
#if HAVE_TDM
    int cur = (kt >> 5) & 1;
    // all ds reads of the other buffer (and pbuf) must be done before reuse
    asm volatile("s_wait_dscnt 0x0" ::: "memory");
    if (kt + 32 < T_) {
      int nxt = cur ^ 1;
      tdm_issue(g1k, kl[nxt], kga + (unsigned long long)(kt + 32) * (DK_ * 2));
      tdm_issue(g1v, vl[nxt], vga + (unsigned long long)(kt + 32) * 2);
      __builtin_amdgcn_s_wait_tensorcnt(2);   // current chunk's 2 DMAs done
    } else {
      __builtin_amdgcn_s_wait_tensorcnt(0);
    }
    const __bf16* kb = kbuf[wv][cur];
    const __bf16* vb = vbuf[wv][cur];
    v8f s0 = {}, s1 = {};
    s0 = wmma_bf(qa0, ldB(kb, 0, DK_, 0, lane), s0);
    s0 = wmma_bf(qa1, ldB(kb, 0, DK_, 32, lane), s0);
    s1 = wmma_bf(qa0, ldB(kb, 16, DK_, 0, lane), s1);
    s1 = wmma_bf(qa1, ldB(kb, 16, DK_, 32, lane), s1);
#else
    v8f s0 = {}, s1 = {};
    s0 = wmma_bf(qa0, ldB(kbase, kt, DK_, 0, lane), s0);
    s0 = wmma_bf(qa1, ldB(kbase, kt, DK_, 32, lane), s0);
    s1 = wmma_bf(qa0, ldB(kbase, kt + 16, DK_, 0, lane), s1);
    s1 = wmma_bf(qa1, ldB(kbase, kt + 16, DK_, 32, lane), s1);
#endif
    bool ka = mrow[kt + col] != 0;
    bool kb2 = mrow[kt + 16 + col] != 0;
    // online softmax per row (rows r+8*hh live in one 16-lane half)
#pragma unroll
    for (int r = 0; r < 8; ++r) {
      float a = ka ? s0[r] : -1e9f;
      float c = kb2 ? s1[r] : -1e9f;
      float tmax = fmaxf(a, c);
#pragma unroll
      for (int mlt = 1; mlt < 16; mlt <<= 1) tmax = fmaxf(tmax, __shfl_xor(tmax, mlt, 32));
      float nm = fmaxf(rmax[r], tmax);
      float scale = __expf(rmax[r] - nm);
      rmax[r] = nm;
      float e0 = __expf(a - nm);
      float e1 = __expf(c - nm);
      float ls = e0 + e1;
#pragma unroll
      for (int mlt = 1; mlt < 16; mlt <<= 1) ls += __shfl_xor(ls, mlt, 32);
      rsum[r] = rsum[r] * scale + ls;
      o0[r] *= scale; o1[r] *= scale; o2[r] *= scale; o3[r] *= scale;
      int m = r + 8 * hh;                 // C-layout -> row-major P in LDS
      pb[m * 32 + col]      = f2bf(e0);
      pb[m * 32 + 16 + col] = f2bf(e1);
    }
    asm volatile("s_wait_dscnt 0x0" ::: "memory");   // P stores visible to wave
    const __bf16* pp = pb + (lane & 15) * 32 + 8 * hh;
    v16bf pa = pack16(pp, pp + 16);                  // P in A-layout (16x32)
#if HAVE_TDM
    o0 = wmma_bf(pa, ldB(vb, 0,  32, 0, lane), o0);
    o1 = wmma_bf(pa, ldB(vb, 16, 32, 0, lane), o1);
    o2 = wmma_bf(pa, ldB(vb, 32, 32, 0, lane), o2);
    o3 = wmma_bf(pa, ldB(vb, 48, 32, 0, lane), o3);
#else
    o0 = wmma_bf(pa, ldB(vbase, 0,  T_, kt, lane), o0);
    o1 = wmma_bf(pa, ldB(vbase, 16, T_, kt, lane), o1);
    o2 = wmma_bf(pa, ldB(vbase, 32, T_, kt, lane), o2);
    o3 = wmma_bf(pa, ldB(vbase, 48, T_, kt, lane), o3);
#endif
  }

#pragma unroll
  for (int r = 0; r < 8; ++r) {
    int m = r + 8 * hh;
    int t = q0 + m;
    float inv = (mrow[t] != 0) ? (1.0f / rsum[r]) : 0.0f;  // query-side mask
    size_t rowb = ((size_t)(b * T_ + t)) * (H_ * DV_) + h * 64;
    O[rowb + col]      = f2bf(o0[r] * inv);
    O[rowb + 16 + col] = f2bf(o1[r] * inv);
    O[rowb + 32 + col] = f2bf(o2[r] * inv);
    O[rowb + 48 + col] = f2bf(o3[r] * inv);
  }
}

// ---------------- Output projection: out = A @ Wo^T + bo (fp32 out) ----------------
__global__ __launch_bounds__(256) void oproj_kernel(
    const __bf16* __restrict__ A, const __bf16* __restrict__ W,
    const float* __restrict__ bias, float* __restrict__ out) {
  int lane = threadIdx.x & 31, wv = threadIdx.x >> 5;
  int g = blockIdx.x * 8 + wv;
  int m0 = (g & 255) * 32;
  int n0 = (g >> 8) * 64;
  v8f acc[2][4] = {};
  for (int k0 = 0; k0 < HID_; k0 += 32) {
    v16bf a0 = ldA(A, m0, HID_, k0, lane);
    v16bf a1 = ldA(A, m0 + 16, HID_, k0, lane);
#pragma unroll
    for (int j = 0; j < 4; ++j) {
      v16bf bm = ldB(W, n0 + 16 * j, HID_, k0, lane);
      acc[0][j] = wmma_bf(a0, bm, acc[0][j]);
      acc[1][j] = wmma_bf(a1, bm, acc[1][j]);
    }
  }
  int col = lane & 15, hh = lane >> 4;
#pragma unroll
  for (int j = 0; j < 4; ++j) {
    int n = n0 + 16 * j + col;
    float bn = bias[n];
#pragma unroll
    for (int i = 0; i < 2; ++i)
#pragma unroll
      for (int r = 0; r < 8; ++r)
        out[(size_t)(m0 + 16 * i + r + 8 * hh) * HID_ + n] = acc[i][j][r] + bn;
  }
}

extern "C" void kernel_launch(void* const* d_in, const int* in_sizes, int n_in,
                              void* d_out, int out_size, void* d_ws, size_t ws_size,
                              hipStream_t stream) {
  (void)in_sizes; (void)n_in; (void)out_size; (void)ws_size;
  const float* query = (const float*)d_in[0];
  const float* key   = (const float*)d_in[1];
  const float* value = (const float*)d_in[2];
  const int*   mask  = (const int*)d_in[3];
  const float* Wq = (const float*)d_in[4];
  const float* bq = (const float*)d_in[5];
  const float* Wk = (const float*)d_in[6];
  const float* bk = (const float*)d_in[7];
  const float* Wv = (const float*)d_in[8];
  const float* bv = (const float*)d_in[9];
  const float* Wo = (const float*)d_in[10];
  const float* bo = (const float*)d_in[11];
  float* out = (float*)d_out;

  const size_t NX = (size_t)B_ * T_ * HID_;   // 8,388,608
  const size_t NW = (size_t)HID_ * HID_;      // 1,048,576
  char* p = (char*)d_ws;
  __bf16* Xq  = (__bf16*)p; p += NX * 2;
  __bf16* Xk  = (__bf16*)p; p += NX * 2;
  __bf16* Xv  = (__bf16*)p; p += NX * 2;
  __bf16* Wqb = (__bf16*)p; p += NW * 2;
  __bf16* Wkb = (__bf16*)p; p += NW * 2;
  __bf16* Wvb = (__bf16*)p; p += NW * 2;
  __bf16* Wob = (__bf16*)p; p += NW * 2;
  __bf16* Qd  = (__bf16*)p; p += NX * 2;      // [b][h][t][dk]
  __bf16* Kd  = (__bf16*)p; p += NX * 2;      // [b][h][t][dk]
  __bf16* VTd = (__bf16*)p; p += NX * 2;      // [b][h][dv][t]
  __bf16* Ad  = (__bf16*)p; p += NX * 2;      // [b][t][h*dv]

  cvt_kernel<<<4096, 256, 0, stream>>>(query, Xq, (int)NX);
  cvt_kernel<<<4096, 256, 0, stream>>>(key,   Xk, (int)NX);
  cvt_kernel<<<4096, 256, 0, stream>>>(value, Xv, (int)NX);
  cvt_kernel<<<1024, 256, 0, stream>>>(Wq, Wqb, (int)NW);
  cvt_kernel<<<1024, 256, 0, stream>>>(Wk, Wkb, (int)NW);
  cvt_kernel<<<1024, 256, 0, stream>>>(Wv, Wvb, (int)NW);
  cvt_kernel<<<1024, 256, 0, stream>>>(Wo, Wob, (int)NW);

  qkv_proj_kernel<<<dim3(512, 1, 3), 256, 0, stream>>>(
      Xq, Xk, Xv, Wqb, Wkb, Wvb, bq, bk, bv, Qd, Kd, VTd);

  attn_kernel<<<(B_ * H_ * (T_ / 16)) / 8, 256, 0, stream>>>(Qd, Kd, VTd, mask, Ad);

  oproj_kernel<<<dim3(512, 1, 1), 256, 0, stream>>>(Ad, Wob, bo, out);
}
